// TransformerEncoder_39857296507555
// MI455X (gfx1250) — matrix operand order repaired and checked by reading, not compiled, not run on hardware
//
#include <hip/hip_runtime.h>

#define DEV __device__ __forceinline__

typedef unsigned short u16;
typedef __bf16 bf16_t;
typedef float        v8f    __attribute__((ext_vector_type(8)));
typedef bf16_t       v16bf  __attribute__((ext_vector_type(16)));
typedef u16          u16x8  __attribute__((ext_vector_type(8)));
typedef u16          u16x16 __attribute__((ext_vector_type(16)));
typedef unsigned int u32x4  __attribute__((ext_vector_type(4)));
typedef int          i32x4  __attribute__((ext_vector_type(4)));
typedef int          i32x8  __attribute__((ext_vector_type(8)));

static constexpr int L_LAYERS = 6;
static constexpr int D_MODEL  = 1024;
static constexpr int N_HEAD   = 16;
static constexpr int D_K      = 64;
static constexpr int D_FF     = 4096;
static constexpr int BATCH    = 4;
static constexpr int SEQ      = 1024;
static constexpr int M_ROWS   = BATCH * SEQ;

DEV u16 f2bf(float f) {
    unsigned u = __builtin_bit_cast(unsigned, f);
    u += 0x7FFFu + ((u >> 16) & 1u);   // round-to-nearest-even
    return (u16)(u >> 16);
}

DEV v16bf cat16(u16x8 lo, u16x8 hi) {
    u16x16 t = __builtin_shufflevector(lo, hi, 0,1,2,3,4,5,6,7,8,9,10,11,12,13,14,15);
    return __builtin_bit_cast(v16bf, t);
}

// LDS byte offset of a generic pointer to __shared__ (flat LDS address truncates
// to the in-allocation offset per CDNA5 aperture mapping, ISA 10.2).
DEV unsigned lds_off(const void* p) { return (unsigned)(size_t)p; }

// TDM: DMA one 2D bf16 tile (tile_y rows x tile_x K-contiguous elems, row
// stride ld elems) from global into LDS at `lds`, padding the LDS destination
// by 4 DWORDs every 16 DWORDs (ISA 8.3/8.4 D#: pad_interval=3, pad_amount=3)
// so each 32-half row lands on a 40-half (80B) LDS pitch.
// Wave-level op (EXEC ignored), tracked on TENSORcnt, in-order per wave.
DEV void tdm_load_tile(unsigned lds, const void* gaddr,
                       int tile_x, int tile_y, int ld)
{
    unsigned long long ga = (unsigned long long)gaddr;
    u32x4 g0;
    g0[0] = 1u;                                        // count=1, user D#
    g0[1] = lds;                                       // lds_addr (bytes)
    g0[2] = (unsigned)(ga & 0xFFFFFFFFu);              // global_addr lo
    g0[3] = (unsigned)((ga >> 32) & 0x01FFFFFFu) | 0x80000000u; // addr hi | type=2
    i32x8 g1;
    g1[0] = (int)((1u << 16)      // data_size = 1 -> 2 bytes
                | (1u << 20)      // pad_enable
                | (3u << 22)      // pad_interval: every 16 DWORDs
                | (3u << 25));    // pad_amount: 4 DWORDs
    g1[1] = (int)(((unsigned)tile_x & 0xFFFFu) << 16);          // tensor_dim0 lo
    g1[2] = (int)(((unsigned)tile_y & 0xFFFFu) << 16);          // dim0 hi=0 | tensor_dim1 lo
    g1[3] = (int)(((unsigned)tile_x & 0xFFFFu) << 16);          // dim1 hi=0 | tile_dim0
    g1[4] = (int)((unsigned)tile_y & 0xFFFFu);                  // tile_dim1 | tile_dim2=0
    g1[5] = (int)(unsigned)ld;                                  // tensor_dim0_stride lo
    g1[6] = 0;                                                  // stride hi | dim1_stride
    g1[7] = 0;
    i32x4 z4 = {0, 0, 0, 0};
#if __clang_major__ >= 23
    i32x8 z8 = {0, 0, 0, 0, 0, 0, 0, 0};
    __builtin_amdgcn_tensor_load_to_lds(g0, g1, z4, z4, z8, 0);
#else
    __builtin_amdgcn_tensor_load_to_lds(g0, g1, z4, z4, 0);
#endif
}

// ---------------------------------------------------------------------------
// Tiled bf16 GEMM core: C[f32] = alpha * A @ B  with A[M][K] row-major (lda)
// and B SUPPLIED K-CONTIGUOUS: B(k,n) = Bt[n*ldb + k]  (N x K layout).
// Both tiles are DMA'd by the Tensor Data Mover: each of the 8 waves issues
// two tensor_load_to_lds per K-step (its BM/8-row slice of A and BN/8-row
// slice of B) into double-buffered LDS, then waits s_wait_tensorcnt so only
// the oldest tile's DMAs are drained while the next tile's stay in flight.
// Fragments are 2 x ds_load_b128 each per the ISA wave32 16-bit A/B striping.
// ---------------------------------------------------------------------------
template<int BM, int BN, int BK, int WM, int WN>
DEV void gemm_core(const u16* __restrict__ A, const u16* __restrict__ Bt,
                   float* __restrict__ C, int K, int lda, int ldb, int ldc,
                   float alpha, int bm, int bn)
{
    constexpr int PAD   = 8;
    constexpr int LDS_K = BK + PAD;            // 40 halves = 80B pitch (16B aligned)
    constexpr int TM    = WM / 16;
    constexpr int TN    = WN / 16;
    constexpr int WAVES_N = BN / WN;
    constexpr int RA = BM / 8;                 // A rows DMA'd per wave
    constexpr int RB = BN / 8;                 // B rows DMA'd per wave

    __shared__ u16 As[2][BM * LDS_K];
    __shared__ u16 Bs[2][BN * LDS_K];

    const int t    = threadIdx.x;
    const int wave = t >> 5;
    const int lane = t & 31;
    const int wm   = (wave / WAVES_N) * WM;
    const int wn   = (wave % WAVES_N) * WN;
    const int l15  = lane & 15;
    const bool hih = (lane & 16) != 0;

    // each wave DMAs its own slice of the A and B tiles (2 TDM ops)
    auto issue = [&](int k0, int buf) {
        tdm_load_tile(lds_off(&As[buf][(wave * RA) * LDS_K]),
                      A + (size_t)(bm + wave * RA) * lda + k0, BK, RA, lda);
        tdm_load_tile(lds_off(&Bs[buf][(wave * RB) * LDS_K]),
                      Bt + (size_t)(bn + wave * RB) * ldb + k0, BK, RB, ldb);
    };

    const v8f vzero = {0.f, 0.f, 0.f, 0.f, 0.f, 0.f, 0.f, 0.f};
    v8f acc[TM][TN];
#pragma unroll
    for (int i = 0; i < TM; ++i)
#pragma unroll
        for (int j = 0; j < TN; ++j)
            acc[i][j] = vzero;

    const int NT = K / BK;
    issue(0, 0);

    for (int kt = 0; kt < NT; ++kt) {
        const int cur = kt & 1;
        if (kt + 1 < NT) {
            issue((kt + 1) * BK, cur ^ 1);
            // per-wave TDM is in-order: oldest tile done when only the 2
            // newly issued ops remain
            __builtin_amdgcn_s_wait_tensorcnt((short)2);
        } else {
            __builtin_amdgcn_s_wait_tensorcnt((short)0);
        }
        __syncthreads();

        // ---- fragment loads (2 x ds_load_b128 each) + WMMA ----
        v16bf afr[TM], bfr[TN];
        const int aoff = hih ? 8 : 0;   // lanes 16-31 start at K=8 (ISA A layout)
#pragma unroll
        for (int i = 0; i < TM; ++i) {
            const u16* p = &As[cur][(wm + i * 16 + l15) * LDS_K + aoff];
            afr[i] = cat16(*(const u16x8*)p, *(const u16x8*)(p + 16));
        }
        const int boff = hih ? 8 : 0;   // lanes 16-31 start at K=8 (same layout as A)
#pragma unroll
        for (int j = 0; j < TN; ++j) {
            const u16* p = &Bs[cur][(wn + j * 16 + l15) * LDS_K + boff];
            bfr[j] = cat16(*(const u16x8*)p, *(const u16x8*)(p + 16));
        }
#pragma unroll
        for (int i = 0; i < TM; ++i)
#pragma unroll
            for (int j = 0; j < TN; ++j)
                acc[i][j] = __builtin_amdgcn_wmma_f32_16x16x32_bf16(
                    false, afr[i], false, bfr[j], (short)0, acc[i][j], false, false);
        __syncthreads();   // all waves done with buf `cur` before it is re-filled
    }

    // ---- epilogue: f32 C layout per ISA (VGPR r -> M=r / 8+r, lane -> N) ----
#pragma unroll
    for (int i = 0; i < TM; ++i) {
#pragma unroll
        for (int j = 0; j < TN; ++j) {
            int col   = bn + wn + j * 16 + l15;
            int rbase = bm + wm + i * 16 + (hih ? 8 : 0);
#pragma unroll
            for (int r = 0; r < 8; ++r)
                C[(size_t)(rbase + r) * ldc + col] = acc[i][j][r] * alpha;
        }
    }
}

// ---------------------------------------------------------------------------
// GEMM wrappers (all B operands are K-contiguous / pre-transposed)
// ---------------------------------------------------------------------------
__global__ __launch_bounds__(256)
void k_gemm(const u16* __restrict__ A, const u16* __restrict__ Bt,
            float* __restrict__ C, int K, int lda, int ldb, int ldc)
{
    gemm_core<256, 128, 32, 64, 64>(A, Bt, C, K, lda, ldb, ldc, 1.0f,
                                    blockIdx.y * 256, blockIdx.x * 128);
}

// scores[b,h] = (Q_h K_h^T)/8 : B(k=d, n=key) = K[key*D + d] is K-contiguous
__global__ __launch_bounds__(256)
void k_scores(const u16* __restrict__ q, const u16* __restrict__ kmat,
              float* __restrict__ s)
{
    int z = blockIdx.z, b = z >> 4, h = z & 15;
    const u16* Ap = q    + (size_t)b * SEQ * D_MODEL + h * D_K;
    const u16* Bp = kmat + (size_t)b * SEQ * D_MODEL + h * D_K;
    float*     Cp = s    + (size_t)z * SEQ * SEQ;
    gemm_core<256, 128, 32, 64, 64>(Ap, Bp, Cp, D_K, D_MODEL, D_MODEL, SEQ,
                                    0.125f, blockIdx.y * 256, blockIdx.x * 128);
}

// O[b,h] = P[b,h] @ V_h with V^T staged as [B][D][S]: B(k,n)=vT[n*S+k]
__global__ __launch_bounds__(256)
void k_av(const u16* __restrict__ probs, const u16* __restrict__ vT,
          float* __restrict__ o)
{
    int z = blockIdx.z, b = z >> 4, h = z & 15;
    const u16* Ap = probs + (size_t)z * SEQ * SEQ;
    const u16* Bp = vT    + (size_t)b * D_MODEL * SEQ + (size_t)(h * D_K) * SEQ;
    float*     Cp = o     + (size_t)b * SEQ * D_MODEL + h * D_K;
    gemm_core<256, 64, 32, 64, 32>(Ap, Bp, Cp, SEQ, SEQ, SEQ, D_MODEL,
                                   1.0f, blockIdx.y * 256, blockIdx.x * 64);
}

// ---------------------------------------------------------------------------
// Elementwise / reduction kernels
// ---------------------------------------------------------------------------
__global__ __launch_bounds__(256)
void k_cast(const float* __restrict__ s, u16* __restrict__ d, size_t n)
{
    for (size_t i = blockIdx.x * 256ull + threadIdx.x; i < n; i += gridDim.x * 256ull)
        d[i] = f2bf(s[i]);
}

// fp32 [L][K][N] weights -> bf16 [L][N][K] (K-contiguous for the TDM GEMM path)
__global__ __launch_bounds__(256)
void k_cast_T(const float* __restrict__ src, u16* __restrict__ dst,
              int K, int N, size_t n)
{
    for (size_t i = blockIdx.x * 256ull + threadIdx.x; i < n; i += gridDim.x * 256ull) {
        size_t kn  = (size_t)K * N;
        size_t l   = i / kn;
        size_t rem = i - l * kn;
        int r = (int)(rem / N), c = (int)(rem - (size_t)r * N);
        dst[l * kn + (size_t)c * K + r] = f2bf(src[i]);
    }
}

__global__ __launch_bounds__(256)
void k_init_x(const float* __restrict__ x, float* __restrict__ xf,
              u16* __restrict__ xb, size_t n)
{
    for (size_t i = blockIdx.x * 256ull + threadIdx.x; i < n; i += gridDim.x * 256ull) {
        float v = x[i];
        xf[i] = v;
        xb[i] = f2bf(v);
    }
}

__global__ __launch_bounds__(256)
void k_bias_cast(const float* __restrict__ y, const float* __restrict__ bias,
                 u16* __restrict__ out, int cols, size_t n)
{
    for (size_t i = blockIdx.x * 256ull + threadIdx.x; i < n; i += gridDim.x * 256ull)
        out[i] = f2bf(y[i] + bias[i & (size_t)(cols - 1)]);
}

// V epilogue: add bias and write V^T as [B][D][S]
__global__ __launch_bounds__(256)
void k_bias_cast_vT(const float* __restrict__ y, const float* __restrict__ bias,
                    u16* __restrict__ vT, size_t n)
{
    for (size_t i = blockIdx.x * 256ull + threadIdx.x; i < n; i += gridDim.x * 256ull) {
        size_t m = i >> 10;               // / D_MODEL
        int    j = (int)(i & 1023);
        size_t b = m >> 10;               // / SEQ
        size_t s = m & 1023;
        vT[b * (size_t)D_MODEL * SEQ + (size_t)j * SEQ + s] = f2bf(y[i] + bias[j]);
    }
}

__global__ __launch_bounds__(256)
void k_bias_relu_cast(const float* __restrict__ y, const float* __restrict__ bias,
                      u16* __restrict__ out, int cols, size_t n)
{
    for (size_t i = blockIdx.x * 256ull + threadIdx.x; i < n; i += gridDim.x * 256ull)
        out[i] = f2bf(fmaxf(y[i] + bias[i & (size_t)(cols - 1)], 0.0f));
}

__global__ __launch_bounds__(256)
void k_bias_residual(const float* __restrict__ y, const float* __restrict__ bias,
                     float* __restrict__ xf, int cols, size_t n)
{
    for (size_t i = blockIdx.x * 256ull + threadIdx.x; i < n; i += gridDim.x * 256ull)
        xf[i] += y[i] + bias[i & (size_t)(cols - 1)];
}

// one block per row of D_MODEL=1024; post-LN; writes fp32 residual + bf16 copy
__global__ __launch_bounds__(256)
void k_layernorm(float* __restrict__ xf, const float* __restrict__ g,
                 const float* __restrict__ be, u16* __restrict__ xb)
{
    __shared__ float red[256];
    const int row = blockIdx.x, t = threadIdx.x;
    float* xr = xf + (size_t)row * D_MODEL;
    u16*   br = xb + (size_t)row * D_MODEL;

    float v[4], s = 0.f;
#pragma unroll
    for (int i = 0; i < 4; ++i) { v[i] = xr[t + i * 256]; s += v[i]; }
    red[t] = s; __syncthreads();
    for (int o = 128; o > 0; o >>= 1) { if (t < o) red[t] += red[t + o]; __syncthreads(); }
    const float mu = red[0] * (1.0f / D_MODEL);
    __syncthreads();

    s = 0.f;
#pragma unroll
    for (int i = 0; i < 4; ++i) { float d = v[i] - mu; s += d * d; }
    red[t] = s; __syncthreads();
    for (int o = 128; o > 0; o >>= 1) { if (t < o) red[t] += red[t + o]; __syncthreads(); }
    const float rstd = rsqrtf(red[0] * (1.0f / D_MODEL) + 1e-5f);
    __syncthreads();

#pragma unroll
    for (int i = 0; i < 4; ++i) {
        int c = t + i * 256;
        float y = (v[i] - mu) * rstd * g[c] + be[c];
        xr[c] = y;
        br[c] = f2bf(y);
    }
}

// one block per score row of SEQ=1024; fp32 softmax -> bf16 probabilities
__global__ __launch_bounds__(256)
void k_softmax(const float* __restrict__ s, u16* __restrict__ p)
{
    __shared__ float red[256];
    const size_t row = blockIdx.x;
    const int t = threadIdx.x;
    const float* sr = s + row * SEQ;
    u16*         pr = p + row * SEQ;

    float v[4], m = -1e30f;
#pragma unroll
    for (int i = 0; i < 4; ++i) { v[i] = sr[t + i * 256]; m = fmaxf(m, v[i]); }
    red[t] = m; __syncthreads();
    for (int o = 128; o > 0; o >>= 1) { if (t < o) red[t] = fmaxf(red[t], red[t + o]); __syncthreads(); }
    m = red[0]; __syncthreads();

    float sum = 0.f;
#pragma unroll
    for (int i = 0; i < 4; ++i) { v[i] = __expf(v[i] - m); sum += v[i]; }
    red[t] = sum; __syncthreads();
    for (int o = 128; o > 0; o >>= 1) { if (t < o) red[t] += red[t + o]; __syncthreads(); }
    const float inv = 1.0f / red[0];

#pragma unroll
    for (int i = 0; i < 4; ++i)
        pr[t + i * 256] = f2bf(v[i] * inv);
}

// ---------------------------------------------------------------------------
// Host orchestration
// ---------------------------------------------------------------------------
extern "C" void kernel_launch(void* const* d_in, const int* in_sizes, int n_in,
                              void* d_out, int out_size, void* d_ws, size_t ws_size,
                              hipStream_t stream)
{
    (void)in_sizes; (void)n_in; (void)out_size; (void)ws_size;

    const float* x   = (const float*)d_in[0];
    const float* Wq  = (const float*)d_in[1];
    const float* bq  = (const float*)d_in[2];
    const float* Wk  = (const float*)d_in[3];
    const float* bk  = (const float*)d_in[4];
    const float* Wv  = (const float*)d_in[5];
    const float* bv  = (const float*)d_in[6];
    const float* Wo  = (const float*)d_in[7];
    const float* bo  = (const float*)d_in[8];
    const float* W1  = (const float*)d_in[9];
    const float* b1  = (const float*)d_in[10];
    const float* W2  = (const float*)d_in[11];
    const float* b2  = (const float*)d_in[12];
    const float* g1  = (const float*)d_in[13];
    const float* be1 = (const float*)d_in[14];
    const float* g2  = (const float*)d_in[15];
    const float* be2 = (const float*)d_in[16];

    constexpr int L = L_LAYERS, D = D_MODEL, F = D_FF, S = SEQ, H = N_HEAD;
    constexpr int Bq = BATCH, Mr = M_ROWS;

    char* ws = (char*)d_ws;
    size_t off = 0;
    auto take = [&](size_t bytes) -> char* {
        char* p = ws + off;
        off += (bytes + 511) & ~(size_t)511;
        return p;
    };
    u16*   wqb = (u16*)take((size_t)L * D * D * 2);   // all weights pre-transposed [N][K]
    u16*   wkb = (u16*)take((size_t)L * D * D * 2);
    u16*   wvb = (u16*)take((size_t)L * D * D * 2);
    u16*   wob = (u16*)take((size_t)L * D * D * 2);
    u16*   w1b = (u16*)take((size_t)L * D * F * 2);
    u16*   w2b = (u16*)take((size_t)L * F * D * 2);
    float* xf  = (float*)take((size_t)Mr * D * 4);
    u16*   xbb = (u16*)take((size_t)Mr * D * 2);
    u16*   qb  = (u16*)take((size_t)Mr * D * 2);
    u16*   kb  = (u16*)take((size_t)Mr * D * 2);
    u16*   vT  = (u16*)take((size_t)Mr * D * 2);      // [B][D][S]
    u16*   ob  = (u16*)take((size_t)Mr * D * 2);
    u16*   hb  = (u16*)take((size_t)Mr * F * 2);
    float* yF  = (float*)take((size_t)Mr * F * 4);
    float* sc  = (float*)take((size_t)Bq * H * S * S * 4);
    u16*   pbuf= (u16*)take((size_t)Bq * H * S * S * 2);

    const dim3 blk(256);
    // weight conversion + transpose to [N][K]
    k_cast_T<<<dim3(4096), blk, 0, stream>>>(Wq, wqb, D, D, (size_t)L * D * D);
    k_cast_T<<<dim3(4096), blk, 0, stream>>>(Wk, wkb, D, D, (size_t)L * D * D);
    k_cast_T<<<dim3(4096), blk, 0, stream>>>(Wv, wvb, D, D, (size_t)L * D * D);
    k_cast_T<<<dim3(4096), blk, 0, stream>>>(Wo, wob, D, D, (size_t)L * D * D);
    k_cast_T<<<dim3(8192), blk, 0, stream>>>(W1, w1b, D, F, (size_t)L * D * F);
    k_cast_T<<<dim3(8192), blk, 0, stream>>>(W2, w2b, F, D, (size_t)L * F * D);
    k_init_x<<<dim3(4096), blk, 0, stream>>>(x, xf, xbb, (size_t)Mr * D);

    for (int l = 0; l < L; ++l) {
        const u16* wq  = wqb + (size_t)l * D * D;
        const u16* wk_ = wkb + (size_t)l * D * D;
        const u16* wv  = wvb + (size_t)l * D * D;
        const u16* wo  = wob + (size_t)l * D * D;
        const u16* w1  = w1b + (size_t)l * D * F;
        const u16* w2  = w2b + (size_t)l * F * D;
        const float *bql = bq + (size_t)l * D, *bkl = bk + (size_t)l * D;
        const float *bvl = bv + (size_t)l * D, *bol = bo + (size_t)l * D;
        const float *b1l = b1 + (size_t)l * F, *b2l = b2 + (size_t)l * D;
        const float *g1l = g1 + (size_t)l * D, *be1l = be1 + (size_t)l * D;
        const float *g2l = g2 + (size_t)l * D, *be2l = be2 + (size_t)l * D;

        const dim3 gD(D / 128, Mr / 256);

        // Q, K, V projections (B = W^T, ldb = K = D)
        k_gemm<<<gD, blk, 0, stream>>>(xbb, wq,  yF, D, D, D, D);
        k_bias_cast<<<dim3(4096), blk, 0, stream>>>(yF, bql, qb, D, (size_t)Mr * D);
        k_gemm<<<gD, blk, 0, stream>>>(xbb, wk_, yF, D, D, D, D);
        k_bias_cast<<<dim3(4096), blk, 0, stream>>>(yF, bkl, kb, D, (size_t)Mr * D);
        k_gemm<<<gD, blk, 0, stream>>>(xbb, wv,  yF, D, D, D, D);
        k_bias_cast_vT<<<dim3(4096), blk, 0, stream>>>(yF, bvl, vT, (size_t)Mr * D);

        // attention
        k_scores<<<dim3(S / 128, S / 256, Bq * H), blk, 0, stream>>>(qb, kb, sc);
        k_softmax<<<dim3(Bq * H * S), blk, 0, stream>>>(sc, pbuf);
        k_av<<<dim3(1, S / 256, Bq * H), blk, 0, stream>>>(pbuf, vT, yF);
        k_cast<<<dim3(4096), blk, 0, stream>>>(yF, ob, (size_t)Mr * D);
        k_gemm<<<gD, blk, 0, stream>>>(ob, wo, yF, D, D, D, D);
        k_bias_residual<<<dim3(4096), blk, 0, stream>>>(yF, bol, xf, D, (size_t)Mr * D);
        k_layernorm<<<dim3(Mr), blk, 0, stream>>>(xf, g1l, be1l, xbb);

        // FFN
        k_gemm<<<dim3(F / 128, Mr / 256), blk, 0, stream>>>(xbb, w1, yF, D, D, D, F);
        k_bias_relu_cast<<<dim3(8192), blk, 0, stream>>>(yF, b1l, hb, F, (size_t)Mr * F);
        k_gemm<<<dim3(D / 128, Mr / 256), blk, 0, stream>>>(hb, w2, yF, F, F, F, D);
        k_bias_residual<<<dim3(4096), blk, 0, stream>>>(yF, b2l, xf, D, (size_t)Mr * D);
        k_layernorm<<<dim3(Mr), blk, 0, stream>>>(xf, g2l, be2l, xbb);
    }

    hipMemcpyAsync(d_out, xf, (size_t)Mr * D * sizeof(float),
                   hipMemcpyDeviceToDevice, stream);
}
